// CompressedModel_21646635171849
// MI455X (gfx1250) — matrix operand order repaired and checked by reading, not compiled, not run on hardware
//
#include <hip/hip_runtime.h>
#include <hip/hip_fp16.h>

typedef __attribute__((ext_vector_type(16))) _Float16 v16h;
typedef __attribute__((ext_vector_type(8)))  float    v8f;
typedef int v4i __attribute__((vector_size(16)));   // raw vector for async-LDS builtin

#define B_DIM   8
#define T_DIM   8192
#define D_DIM   256
#define HALF_T  4096                 // tokens per parity
#define NTILES  256                  // HALF_T / 16
#define KSTEPS  8                    // 256 / 32
#define R_MERGE 409                  // floor(T - T*0.95), capped at T/2
#define UNM     (HALF_T - R_MERGE)   // 3687
#define OUT_T   (UNM + HALF_T)       // 7783

#define AS1 __attribute__((address_space(1)))
#define AS3 __attribute__((address_space(3)))

#if defined(__gfx1250__) && __has_builtin(__builtin_amdgcn_global_load_async_to_lds_b128)
#define HAVE_ASYNC_LDS 1
#else
#define HAVE_ASYNC_LDS 0
#endif

__device__ __forceinline__ void wait_asynccnt0() {
#if __has_builtin(__builtin_amdgcn_s_wait_asynccnt)
    __builtin_amdgcn_s_wait_asynccnt(0);
#else
    asm volatile("s_wait_asynccnt 0x0" ::: "memory");
#endif
}

// ---------------------------------------------------------------------------
// Kernel 1: L2-normalize each token (f32 -> f16) and pack straight into the
// WMMA fragment layouts (ISA 7.12.2).
//   A (16-bit 16x32): lane m+16g holds K = kstep*32 + g*8 + {0..7} and {16..23}
//   B (16-bit 32x16): lane n+16g holds K = kstep*32 + g*8 + {0..7} and {16..23}
// One wave per token; lane L holds the 8 contiguous channels K = 8L..8L+7.
// ---------------------------------------------------------------------------
__global__ void __launch_bounds__(256)
normalize_pack(const float* __restrict__ x,
               _Float16* __restrict__ apk,
               _Float16* __restrict__ bpk)
{
    const int wave = (blockIdx.x * blockDim.x + threadIdx.x) >> 5;  // token id
    const int lane = threadIdx.x & 31;
    const int b = wave / T_DIM;
    const int t = wave % T_DIM;

    const float* src = x + (size_t)wave * D_DIM + lane * 8;
    float4 f0 = *(const float4*)(src);
    float4 f1 = *(const float4*)(src + 4);

    float ss = f0.x*f0.x + f0.y*f0.y + f0.z*f0.z + f0.w*f0.w
             + f1.x*f1.x + f1.y*f1.y + f1.z*f1.z + f1.w*f1.w;
    #pragma unroll
    for (int off = 16; off > 0; off >>= 1) ss += __shfl_xor(ss, off, 32);
    const float inv = 1.0f / fmaxf(sqrtf(ss), 1e-12f);

    union { _Float16 h[8]; uint4 u; } pk;
    pk.h[0] = (_Float16)(f0.x * inv); pk.h[1] = (_Float16)(f0.y * inv);
    pk.h[2] = (_Float16)(f0.z * inv); pk.h[3] = (_Float16)(f0.w * inv);
    pk.h[4] = (_Float16)(f1.x * inv); pk.h[5] = (_Float16)(f1.y * inv);
    pk.h[6] = (_Float16)(f1.z * inv); pk.h[7] = (_Float16)(f1.w * inv);

    const int i    = t >> 1;          // index within parity stream
    const int tile = i >> 4;
    const int m    = i & 15;
    const int c    = lane;            // K-chunk id: K = 8c..8c+7
    const int kstep = c >> 2;

    if ((t & 1) == 0) {               // even token -> A panel
        const int g  = c & 1;
        const int hg = (c >> 1) & 1;
        const size_t o = ((((size_t)b * NTILES + tile) * KSTEPS + kstep) * 32
                          + (m + 16 * g)) * 16 + hg * 8;
        *(uint4*)(apk + o) = pk.u;
    } else {                          // odd token -> B panel
        const int g  = c & 1;
        const int hh = (c >> 1) & 1;
        const size_t o = ((((size_t)b * NTILES + tile) * KSTEPS + kstep) * 32
                          + (m + 16 * g)) * 16 + hh * 8;
        *(uint4*)(bpk + o) = pk.u;
    }
}

// ---------------------------------------------------------------------------
// Kernel 2: fused scores-GEMM + row max/argmax.
// Block = 8 waves; wave w owns TWO 16-row M-tiles (two independent WMMA
// accumulator chains per B fragment -> 2x matrix-pipe ILP, 2x FLOPs per LDS
// byte). B tile (8KB) double-buffered in LDS via CDNA5 async-to-LDS loads
// (ASYNCcnt), one barrier per N-step.
// ---------------------------------------------------------------------------
__global__ void __launch_bounds__(256)
gemm_argmax(const _Float16* __restrict__ apk,
            const _Float16* __restrict__ bpk,
            float* __restrict__ node_max,
            int*   __restrict__ node_idx)
{
    __shared__ __align__(16) _Float16 lds[2 * KSTEPS * 32 * 16];  // 2 x 8 KB

    const int b      = blockIdx.x >> 4;         // 16 blocks per batch
    const int mblk   = blockIdx.x & 15;
    const int w      = threadIdx.x >> 5;
    const int lane   = threadIdx.x & 31;
    const int mtile0 = mblk * 16 + w * 2;       // 256 rows per block
    const int mtile1 = mtile0 + 1;

    // Resident A fragments for both M-tiles (32B coalesced per lane each)
    v16h af0[KSTEPS], af1[KSTEPS];
    {
        const _Float16* a0 =
            apk + ((((size_t)b * NTILES + mtile0) * KSTEPS) * 32 + lane) * 16;
        const _Float16* a1 =
            apk + ((((size_t)b * NTILES + mtile1) * KSTEPS) * 32 + lane) * 16;
        #pragma unroll
        for (int k = 0; k < KSTEPS; ++k) {
            af0[k] = *(const v16h*)(a0 + (size_t)k * 512);
            af1[k] = *(const v16h*)(a1 + (size_t)k * 512);
        }
    }

    float rmax0[8], rmax1[8];
    int   ridx0[8], ridx1[8];
    #pragma unroll
    for (int i = 0; i < 8; ++i) {
        rmax0[i] = -3.4e38f; ridx0[i] = 0;
        rmax1[i] = -3.4e38f; ridx1[i] = 0;
    }

    const uint4* bsrc = (const uint4*)(bpk + (size_t)b * NTILES * KSTEPS * 512);
    uint4* ldsq = (uint4*)lds;                  // 512 uint4 per buffer

#if HAVE_ASYNC_LDS
    // ---- async double-buffered pipeline: 1 barrier per N-step ----
    {   // prefetch tile 0 into buffer 0
        const uint4* s = bsrc;
        __builtin_amdgcn_global_load_async_to_lds_b128(
            (AS1 v4i*)(s + threadIdx.x),
            (AS3 v4i*)(ldsq + threadIdx.x), 0, 0);
        __builtin_amdgcn_global_load_async_to_lds_b128(
            (AS1 v4i*)(s + threadIdx.x + 256),
            (AS3 v4i*)(ldsq + threadIdx.x + 256), 0, 0);
    }
    for (int nt = 0; nt < NTILES; ++nt) {
        const int cur = nt & 1;
        wait_asynccnt0();
        __syncthreads();                        // buf[cur] resident block-wide
        if (nt + 1 < NTILES) {                  // kick off next tile
            const uint4* s = bsrc + (size_t)(nt + 1) * 512;
            uint4* d = ldsq + (1 - cur) * 512;
            __builtin_amdgcn_global_load_async_to_lds_b128(
                (AS1 v4i*)(s + threadIdx.x),
                (AS3 v4i*)(d + threadIdx.x), 0, 0);
            __builtin_amdgcn_global_load_async_to_lds_b128(
                (AS1 v4i*)(s + threadIdx.x + 256),
                (AS3 v4i*)(d + threadIdx.x + 256), 0, 0);
        }
        const _Float16* buf = lds + cur * 4096;
        v8f acc0 = {}, acc1 = {};
        #pragma unroll
        for (int k = 0; k < KSTEPS; ++k) {
            v16h bf = *(const v16h*)(buf + (k * 32 + lane) * 16);
            acc0 = __builtin_amdgcn_wmma_f32_16x16x32_f16(
                       false, af0[k], false, bf, (short)0, acc0, false, false);
            acc1 = __builtin_amdgcn_wmma_f32_16x16x32_f16(
                       false, af1[k], false, bf, (short)0, acc1, false, false);
        }
        const int col = nt * 16 + (lane & 15);
        #pragma unroll
        for (int i = 0; i < 8; ++i) {
            if (acc0[i] > rmax0[i]) { rmax0[i] = acc0[i]; ridx0[i] = col; }
            if (acc1[i] > rmax1[i]) { rmax1[i] = acc1[i]; ridx1[i] = col; }
        }
    }
#else
    // ---- fallback: register-prefetch, single buffer ----
    uint4 r0 = bsrc[threadIdx.x];
    uint4 r1 = bsrc[threadIdx.x + 256];
    for (int nt = 0; nt < NTILES; ++nt) {
        ldsq[threadIdx.x]       = r0;
        ldsq[threadIdx.x + 256] = r1;
        __syncthreads();
        if (nt + 1 < NTILES) {
            const uint4* s = bsrc + (size_t)(nt + 1) * 512;
            r0 = s[threadIdx.x];
            r1 = s[threadIdx.x + 256];
        }
        v8f acc0 = {}, acc1 = {};
        #pragma unroll
        for (int k = 0; k < KSTEPS; ++k) {
            v16h bf = *(const v16h*)(lds + (k * 32 + lane) * 16);
            acc0 = __builtin_amdgcn_wmma_f32_16x16x32_f16(
                       false, af0[k], false, bf, (short)0, acc0, false, false);
            acc1 = __builtin_amdgcn_wmma_f32_16x16x32_f16(
                       false, af1[k], false, bf, (short)0, acc1, false, false);
        }
        const int col = nt * 16 + (lane & 15);
        #pragma unroll
        for (int i = 0; i < 8; ++i) {
            if (acc0[i] > rmax0[i]) { rmax0[i] = acc0[i]; ridx0[i] = col; }
            if (acc1[i] > rmax1[i]) { rmax1[i] = acc1[i]; ridx1[i] = col; }
        }
        __syncthreads();
    }
#endif

    // reduce (max, lowest-index-on-tie) across the 16 lanes of each half-wave
    #pragma unroll
    for (int i = 0; i < 8; ++i) {
        float v0 = rmax0[i]; int id0 = ridx0[i];
        float v1 = rmax1[i]; int id1 = ridx1[i];
        #pragma unroll
        for (int off = 1; off < 16; off <<= 1) {
            float ov0 = __shfl_xor(v0, off, 32);
            int   oi0 = __shfl_xor(id0, off, 32);
            float ov1 = __shfl_xor(v1, off, 32);
            int   oi1 = __shfl_xor(id1, off, 32);
            if (ov0 > v0 || (ov0 == v0 && oi0 < id0)) { v0 = ov0; id0 = oi0; }
            if (ov1 > v1 || (ov1 == v1 && oi1 < id1)) { v1 = ov1; id1 = oi1; }
        }
        rmax0[i] = v0; ridx0[i] = id0;
        rmax1[i] = v1; ridx1[i] = id1;
    }

    const int half = (lane >= 16) ? 8 : 0;
    if ((lane & 15) == 0) {
        #pragma unroll
        for (int i = 0; i < 8; ++i) {
            node_max[b * HALF_T + mtile0 * 16 + half + i] = rmax0[i];
            node_idx[b * HALF_T + mtile0 * 16 + half + i] = ridx0[i];
            node_max[b * HALF_T + mtile1 * 16 + half + i] = rmax1[i];
            node_idx[b * HALF_T + mtile1 * 16 + half + i] = ridx1[i];
        }
    }
}

// ---------------------------------------------------------------------------
// Kernel 3: descending bitonic argsort of node_max (4096 per batch) in LDS,
// stable-tie emulation of jnp.argsort(-node_max).
// ---------------------------------------------------------------------------
__global__ void __launch_bounds__(1024)
sort_edges(const float* __restrict__ node_max, int* __restrict__ edge_idx)
{
    __shared__ float vs[HALF_T];
    __shared__ int   ids[HALF_T];
    const int b = blockIdx.x;

    for (int i = threadIdx.x; i < HALF_T; i += 1024) {
        vs[i]  = node_max[b * HALF_T + i];
        ids[i] = i;
    }
    __syncthreads();

    for (int k = 2; k <= HALF_T; k <<= 1) {
        for (int j = k >> 1; j > 0; j >>= 1) {
            for (int i = threadIdx.x; i < HALF_T; i += 1024) {
                const int ixj = i ^ j;
                if (ixj > i) {
                    const bool desc = ((i & k) == 0);
                    float a = vs[i], c = vs[ixj];
                    int  ai = ids[i], ci = ids[ixj];
                    const bool swp = desc ? (a < c || (a == c && ai > ci))
                                          : (a > c || (a == c && ai < ci));
                    if (swp) { vs[i] = c; vs[ixj] = a; ids[i] = ci; ids[ixj] = ai; }
                }
            }
            __syncthreads();
        }
    }

    for (int i = threadIdx.x; i < HALF_T; i += 1024)
        edge_idx[b * HALF_T + i] = ids[i];
}

// ---------------------------------------------------------------------------
// Kernel 4a: gather the unmerged (kept) even tokens to the output.
// ---------------------------------------------------------------------------
__global__ void __launch_bounds__(256)
gather_unm(const float* __restrict__ x, const int* __restrict__ edge_idx,
           float* __restrict__ out)
{
    const int b = blockIdx.y;
    const int j = blockIdx.x;                  // 0..UNM-1
    const int d = threadIdx.x;
    const int row = edge_idx[b * HALF_T + R_MERGE + j];
    out[((size_t)b * OUT_T + j) * D_DIM + d] =
        x[((size_t)b * T_DIM + 2 * row) * D_DIM + d];
}

// ---------------------------------------------------------------------------
// Kernel 4b: deterministic scatter-mean merge (no float atomics): each dst row
// scans the r merge entries cached in LDS and accumulates in fixed order.
// ---------------------------------------------------------------------------
__global__ void __launch_bounds__(256)
merge_dst(const float* __restrict__ x, const int* __restrict__ edge_idx,
          const int* __restrict__ node_idx, float* __restrict__ out)
{
    __shared__ int sSrc[R_MERGE];
    __shared__ int sDst[R_MERGE];
    const int b = blockIdx.y;
    const int i = blockIdx.x;                  // dst row 0..HALF_T-1

    for (int s = threadIdx.x; s < R_MERGE; s += 256) {
        const int sr = edge_idx[b * HALF_T + s];
        sSrc[s] = sr;
        sDst[s] = node_idx[b * HALF_T + sr];
    }
    __syncthreads();

    const int d = threadIdx.x;
    float val = x[((size_t)b * T_DIM + 2 * i + 1) * D_DIM + d];
    float cnt = 0.0f;
    for (int s = 0; s < R_MERGE; ++s) {
        if (sDst[s] == i) {
            val += x[((size_t)b * T_DIM + 2 * sSrc[s]) * D_DIM + d];
            cnt += 1.0f;
        }
    }
    out[((size_t)b * OUT_T + UNM + i) * D_DIM + d] = val / (1.0f + cnt);
}

// ---------------------------------------------------------------------------
extern "C" void kernel_launch(void* const* d_in, const int* in_sizes, int n_in,
                              void* d_out, int out_size, void* d_ws, size_t ws_size,
                              hipStream_t stream)
{
    (void)in_sizes; (void)n_in; (void)out_size; (void)ws_size;
    const float* x = (const float*)d_in[0];
    float* out = (float*)d_out;

    // workspace layout (~32.4 MB)
    char* ws = (char*)d_ws;
    const size_t PK_BYTES = (size_t)B_DIM * NTILES * KSTEPS * 32 * 16 * 2; // 16 MB
    _Float16* apk      = (_Float16*)(ws);
    _Float16* bpk      = (_Float16*)(ws + PK_BYTES);
    float*    node_max = (float*)(ws + 2 * PK_BYTES);
    int*      node_idx = (int*)  (ws + 2 * PK_BYTES + (size_t)B_DIM * HALF_T * 4);
    int*      edge_idx = (int*)  (ws + 2 * PK_BYTES + (size_t)B_DIM * HALF_T * 8);

    // 1) normalize + fragment-pack: one wave per token, 8 waves per block
    normalize_pack<<<(B_DIM * T_DIM) / 8, 256, 0, stream>>>(x, apk, bpk);

    // 2) fused WMMA GEMM + row max/argmax: 16 blocks per batch (256 rows each)
    gemm_argmax<<<B_DIM * 16, 256, 0, stream>>>(apk, bpk, node_max, node_idx);

    // 3) per-batch bitonic argsort
    sort_edges<<<B_DIM, 1024, 0, stream>>>(node_max, edge_idx);

    // 4) outputs
    gather_unm<<<dim3(UNM, B_DIM), 256, 0, stream>>>(x, edge_idx, out);
    merge_dst<<<dim3(HALF_T, B_DIM), 256, 0, stream>>>(x, edge_idx, node_idx, out);
}
